// QKVAttention_12017318494262
// MI455X (gfx1250) — compile-verified, matrix operand order checked by reading, not run on hardware
//
#include <hip/hip_runtime.h>
#include <hip/hip_bf16.h>

// ---------------------------------------------------------------------------
// MI455X (gfx1250) fused multi-head attention, f16 WMMA path.
//   x:[4,16,512,256] f32 -> out:[4,16,512,512] f32
// Pipeline: cvt(f32->f16) -> GEMM(Q,K,V) -> flash-attention -> GEMM(proj)+bias
// All matmuls use v_wmma_f32_16x16x32_f16 (wave32), LDS double-buffered.
// ---------------------------------------------------------------------------

typedef __attribute__((ext_vector_type(16))) _Float16 v16h;
typedef __attribute__((ext_vector_type(8)))  _Float16 v8h;
typedef __attribute__((ext_vector_type(4)))  _Float16 v4h;
typedef __attribute__((ext_vector_type(8)))  float    v8f;

#define WMMA_F16(a, b, c) \
  __builtin_amdgcn_wmma_f32_16x16x32_f16(false, (a), false, (b), (short)0, (c), false, false)

// Problem constants
#define BN   64      // B*N
#define Tq   512     // sequence length
#define Cd   256     // input channels
#define Hd   512     // hidden
#define NH   8       // heads
#define HD   64      // per-head dim
#define MROWS 32768  // BN * Tq

// Workspace layout (bytes)
#define OFF_XH   ((size_t)0)
#define OFF_QH   ((size_t)16777216)
#define OFF_KH   ((size_t)50331648)
#define OFF_VH   ((size_t)83886080)
#define OFF_HOH  ((size_t)117440512)
#define OFF_WQH  ((size_t)150994944)
#define OFF_WKH  ((size_t)151257088)
#define OFF_WVH  ((size_t)151519232)
#define OFF_WPH  ((size_t)151781376)

// Build a 16x32 f16 A/B fragment from a padded-row buffer.
// Per CDNA5 ISA: lanes 0-15 take K elements {koff..+7, koff+16..+23},
// lanes 16-31 take {koff+8.., koff+24..} -> caller passes koff incl. (half?8:0).
__device__ __forceinline__ v16h make_frag(const _Float16* p, int koff) {
  v8h lo = *(const v8h*)(p + koff);
  v8h hi = *(const v8h*)(p + koff + 16);
  v16h r;
#pragma unroll
  for (int i = 0; i < 8; ++i) { r[i] = lo[i]; r[8 + i] = hi[i]; }
  return r;
}

// ---------------------------------------------------------------------------
// f32 -> f16 conversion (n divisible by 4)
// ---------------------------------------------------------------------------
__global__ void cvt_f32_f16(const float* __restrict__ s, _Float16* __restrict__ d, int n) {
  int i = (blockIdx.x * blockDim.x + threadIdx.x) * 4;
  int stride = gridDim.x * blockDim.x * 4;
  for (; i < n; i += stride) {
    float4 f = *(const float4*)(s + i);
    v4h h = { (_Float16)f.x, (_Float16)f.y, (_Float16)f.z, (_Float16)f.w };
    *(v4h*)(d + i) = h;
  }
}

// ---------------------------------------------------------------------------
// Tiled WMMA GEMM: C[M x N] = A[M x K] * B[K x N]
// WG = 256 thr (8 waves). WG tile 128x64, wave tile 16x64, K-step 32.
// LDS double-buffered; one barrier per K-step.
// outH != nullptr -> f16 output; else f32 output + bias.
// ---------------------------------------------------------------------------
__global__ __launch_bounds__(256)
void gemm_f16_wmma(const _Float16* __restrict__ A, int lda,
                   const _Float16* __restrict__ B, int ldb, int K,
                   _Float16* __restrict__ outH, float* __restrict__ outF,
                   const float* __restrict__ bias, int ldo) {
  __shared__ _Float16 As[2][128 * 40];  // 128 x 32, rows padded to 40
  __shared__ _Float16 Bt[2][64 * 40];   // B tile transposed: [n][k]

  const int tid  = threadIdx.x;
  const int wave = tid >> 5;
  const int lane = tid & 31;
  const int l15  = lane & 15;
  const int half = lane >> 4;
  const int kb   = half ? 8 : 0;

  const int m0 = blockIdx.x * 128;
  const int n0 = blockIdx.y * 64;

  // coop-load geometry
  const int arow0 = tid >> 2, ach0 = (tid & 3) * 8;          // A chunk 0
  const int arow1 = (tid + 256) >> 2, ach1 = ach0;           // A chunk 1
  const int brow  = tid >> 3, bch = (tid & 7) * 8;           // B chunk

  v8f acc[4];
  const v8f zero = {0.f, 0.f, 0.f, 0.f, 0.f, 0.f, 0.f, 0.f};
#pragma unroll
  for (int i = 0; i < 4; ++i) acc[i] = zero;

  // prologue: stage k0 = 0 into buffer 0
  {
    v8h a0 = *(const v8h*)(A + (size_t)(m0 + arow0) * lda + ach0);
    v8h a1 = *(const v8h*)(A + (size_t)(m0 + arow1) * lda + ach1);
    v8h b0 = *(const v8h*)(B + (size_t)brow * ldb + n0 + bch);
    *(v8h*)(&As[0][arow0 * 40 + ach0]) = a0;
    *(v8h*)(&As[0][arow1 * 40 + ach1]) = a1;
#pragma unroll
    for (int j = 0; j < 8; ++j) Bt[0][(bch + j) * 40 + brow] = b0[j];
  }
  __syncthreads();

  const int nsteps = K >> 5;
  for (int it = 0; it < nsteps; ++it) {
    const int cur = it & 1, nxt = cur ^ 1;
    const bool more = (it + 1) < nsteps;
    const int k1 = (it + 1) << 5;

    // issue next tile's global loads early (overlap with WMMA below)
    v8h na0, na1, nb0;
    if (more) {
      na0 = *(const v8h*)(A + (size_t)(m0 + arow0) * lda + k1 + ach0);
      na1 = *(const v8h*)(A + (size_t)(m0 + arow1) * lda + k1 + ach1);
      nb0 = *(const v8h*)(B + (size_t)(k1 + brow) * ldb + n0 + bch);
    }

    // hoist all fragments, then issue 4 back-to-back WMMAs
    v16h af = make_frag(&As[cur][(wave * 16 + l15) * 40], kb);
    v16h bf0 = make_frag(&Bt[cur][(0 * 16 + l15) * 40], kb);
    v16h bf1 = make_frag(&Bt[cur][(1 * 16 + l15) * 40], kb);
    v16h bf2 = make_frag(&Bt[cur][(2 * 16 + l15) * 40], kb);
    v16h bf3 = make_frag(&Bt[cur][(3 * 16 + l15) * 40], kb);
    acc[0] = WMMA_F16(af, bf0, acc[0]);
    acc[1] = WMMA_F16(af, bf1, acc[1]);
    acc[2] = WMMA_F16(af, bf2, acc[2]);
    acc[3] = WMMA_F16(af, bf3, acc[3]);

    if (more) {
      *(v8h*)(&As[nxt][arow0 * 40 + ach0]) = na0;
      *(v8h*)(&As[nxt][arow1 * 40 + ach1]) = na1;
#pragma unroll
      for (int j = 0; j < 8; ++j) Bt[nxt][(bch + j) * 40 + brow] = nb0[j];
    }
    __syncthreads();
  }

  // Epilogue (C layout: VGPR r -> row r / r+8 per lane half, col = l15)
#pragma unroll
  for (int nf = 0; nf < 4; ++nf) {
#pragma unroll
    for (int r = 0; r < 8; ++r) {
      int row = m0 + wave * 16 + r + (half ? 8 : 0);
      int col = n0 + nf * 16 + l15;
      float v = acc[nf][r];
      if (outH) outH[(size_t)row * ldo + col] = (_Float16)v;
      else      outF[(size_t)row * ldo + col] = v + bias[col];
    }
  }
}

// ---------------------------------------------------------------------------
// Flash attention over T=512, per (b*n, head). WG = 256 thr (8 waves).
// Wave owns 16 query rows; WG owns 128. grid = (T/128, BN*NH).
// K/V tiles double-buffered in LDS.
// ---------------------------------------------------------------------------
__global__ __launch_bounds__(256)
void attn_wmma(const _Float16* __restrict__ qm, const _Float16* __restrict__ km,
               const _Float16* __restrict__ vm, _Float16* __restrict__ om) {
  __shared__ _Float16 Ks[2][32 * 72];     // K tile: [key][d], padded
  __shared__ _Float16 Vt[2][64 * 40];     // V tile transposed: [d][key], padded
  __shared__ _Float16 Pb[8 * 16 * 40];    // per-wave P (16x32), padded

  const int tid  = threadIdx.x;
  const int wave = tid >> 5;
  const int lane = tid & 31;
  const int l15  = lane & 15;
  const int half = lane >> 4;
  const int kb   = half ? 8 : 0;

  const int bn = blockIdx.y >> 3;
  const int hh = blockIdx.y & 7;
  const size_t rowbase = (size_t)bn * Tq;
  const int q0 = blockIdx.x * 128 + wave * 16;
  const int hcol = hh * HD;
  const float scale = 0.125f;  // 1/sqrt(64)

  _Float16* Pw = &Pb[wave * 16 * 40];

  // coop-load geometry for 32x64 K/V tiles
  const int cr = tid >> 3, cc = (tid & 7) * 8;

  // Q fragments (16 rows x 64 d), resident for whole kernel
  v16h aq[2];
#pragma unroll
  for (int kk = 0; kk < 2; ++kk)
    aq[kk] = make_frag(qm + (rowbase + q0 + l15) * (size_t)Hd + hcol + kk * 32, kb);

  float m[8], l[8];
  v8f o[4];
  const v8f zero = {0.f, 0.f, 0.f, 0.f, 0.f, 0.f, 0.f, 0.f};
#pragma unroll
  for (int r = 0; r < 8; ++r) { m[r] = -3.4e38f; l[r] = 0.f; }
#pragma unroll
  for (int t = 0; t < 4; ++t) o[t] = zero;

  // prologue: stage kt = 0 into buffer 0
  {
    size_t g = (rowbase + cr) * (size_t)Hd + hcol + cc;
    v8h kv = *(const v8h*)(km + g);
    v8h vv = *(const v8h*)(vm + g);
    *(v8h*)(&Ks[0][cr * 72 + cc]) = kv;
#pragma unroll
    for (int j = 0; j < 8; ++j) Vt[0][(cc + j) * 40 + cr] = vv[j];
  }
  __syncthreads();

  for (int kt = 0; kt < Tq / 32; ++kt) {
    const int cur = kt & 1, nxt = cur ^ 1;
    const bool more = (kt + 1) < Tq / 32;

    // issue next K/V tile global loads early
    v8h nk, nv;
    if (more) {
      size_t g = (rowbase + (kt + 1) * 32 + cr) * (size_t)Hd + hcol + cc;
      nk = *(const v8h*)(km + g);
      nv = *(const v8h*)(vm + g);
    }

    // S = Q * K^T : hoist 4 B-fragments, then 4 WMMAs
    v16h bk00 = make_frag(&Ks[cur][(0 * 16 + l15) * 72], 0 * 32 + kb);
    v16h bk01 = make_frag(&Ks[cur][(0 * 16 + l15) * 72], 1 * 32 + kb);
    v16h bk10 = make_frag(&Ks[cur][(1 * 16 + l15) * 72], 0 * 32 + kb);
    v16h bk11 = make_frag(&Ks[cur][(1 * 16 + l15) * 72], 1 * 32 + kb);
    v8f s0 = zero, s1 = zero;
    s0 = WMMA_F16(aq[0], bk00, s0);
    s1 = WMMA_F16(aq[0], bk10, s1);
    s0 = WMMA_F16(aq[1], bk01, s0);
    s1 = WMMA_F16(aq[1], bk11, s1);

    // Online softmax update (C layout rows)
#pragma unroll
    for (int r = 0; r < 8; ++r) {
      float x0 = s0[r] * scale, x1 = s1[r] * scale;
      float tmax = fmaxf(x0, x1);
#pragma unroll
      for (int msk = 1; msk < 16; msk <<= 1) tmax = fmaxf(tmax, __shfl_xor(tmax, msk));
      float mnew = fmaxf(m[r], tmax);
      float fac  = __expf(m[r] - mnew);
      float p0   = __expf(x0 - mnew);
      float p1   = __expf(x1 - mnew);
      float rs   = p0 + p1;
#pragma unroll
      for (int msk = 1; msk < 16; msk <<= 1) rs += __shfl_xor(rs, msk);
      l[r] = l[r] * fac + rs;
      m[r] = mnew;
#pragma unroll
      for (int t = 0; t < 4; ++t) o[t][r] *= fac;
      int prow = r + (half ? 8 : 0);
      Pw[prow * 40 + l15]      = (_Float16)p0;
      Pw[prow * 40 + 16 + l15] = (_Float16)p1;
    }

    // O += P * V : hoist A-frag + 4 B-frags, then 4 WMMAs
    v16h ap  = make_frag(&Pw[l15 * 40], kb);
    v16h bv0 = make_frag(&Vt[cur][(0 * 16 + l15) * 40], kb);
    v16h bv1 = make_frag(&Vt[cur][(1 * 16 + l15) * 40], kb);
    v16h bv2 = make_frag(&Vt[cur][(2 * 16 + l15) * 40], kb);
    v16h bv3 = make_frag(&Vt[cur][(3 * 16 + l15) * 40], kb);
    o[0] = WMMA_F16(ap, bv0, o[0]);
    o[1] = WMMA_F16(ap, bv1, o[1]);
    o[2] = WMMA_F16(ap, bv2, o[2]);
    o[3] = WMMA_F16(ap, bv3, o[3]);

    // store next tile to the other buffer, single barrier per iteration
    if (more) {
      *(v8h*)(&Ks[nxt][cr * 72 + cc]) = nk;
#pragma unroll
      for (int j = 0; j < 8; ++j) Vt[nxt][(cc + j) * 40 + cr] = nv[j];
    }
    __syncthreads();
  }

  // Normalize and write head_out (f16)
#pragma unroll
  for (int r = 0; r < 8; ++r) {
    float inv = 1.f / l[r];
    int row = q0 + r + (half ? 8 : 0);
#pragma unroll
    for (int t = 0; t < 4; ++t) {
      int col = hcol + t * 16 + l15;
      om[(rowbase + row) * (size_t)Hd + col] = (_Float16)(o[t][r] * inv);
    }
  }
}

// ---------------------------------------------------------------------------
extern "C" void kernel_launch(void* const* d_in, const int* in_sizes, int n_in,
                              void* d_out, int out_size, void* d_ws, size_t ws_size,
                              hipStream_t stream) {
  const float* x  = (const float*)d_in[0];
  const float* Wq = (const float*)d_in[1];
  const float* Wk = (const float*)d_in[2];
  const float* Wv = (const float*)d_in[3];
  const float* Wp = (const float*)d_in[4];
  const float* bp = (const float*)d_in[5];
  float* out = (float*)d_out;

  char* ws = (char*)d_ws;
  _Float16* xh  = (_Float16*)(ws + OFF_XH);
  _Float16* qh  = (_Float16*)(ws + OFF_QH);
  _Float16* kh  = (_Float16*)(ws + OFF_KH);
  _Float16* vh  = (_Float16*)(ws + OFF_VH);
  _Float16* hoh = (_Float16*)(ws + OFF_HOH);
  _Float16* wqh = (_Float16*)(ws + OFF_WQH);
  _Float16* wkh = (_Float16*)(ws + OFF_WKH);
  _Float16* wvh = (_Float16*)(ws + OFF_WVH);
  _Float16* wph = (_Float16*)(ws + OFF_WPH);

  // 1) f32 -> f16 conversions
  cvt_f32_f16<<<2048, 256, 0, stream>>>(x,  xh,  MROWS * Cd);
  cvt_f32_f16<<<128,  256, 0, stream>>>(Wq, wqh, Cd * Hd);
  cvt_f32_f16<<<128,  256, 0, stream>>>(Wk, wkh, Cd * Hd);
  cvt_f32_f16<<<128,  256, 0, stream>>>(Wv, wvh, Cd * Hd);
  cvt_f32_f16<<<256,  256, 0, stream>>>(Wp, wph, Hd * Hd);

  // 2) Q/K/V projections: [32768 x 256] x [256 x 512] -> f16
  dim3 gq(MROWS / 128, Hd / 64);
  gemm_f16_wmma<<<gq, 256, 0, stream>>>(xh, Cd, wqh, Hd, Cd, qh, nullptr, nullptr, Hd);
  gemm_f16_wmma<<<gq, 256, 0, stream>>>(xh, Cd, wkh, Hd, Cd, kh, nullptr, nullptr, Hd);
  gemm_f16_wmma<<<gq, 256, 0, stream>>>(xh, Cd, wvh, Hd, Cd, vh, nullptr, nullptr, Hd);

  // 3) Flash attention per (bn, head)
  dim3 ga(Tq / 128, BN * NH);
  attn_wmma<<<ga, 256, 0, stream>>>(qh, kh, vh, hoh);

  // 4) Output projection + bias: [32768 x 512] x [512 x 512] -> f32
  gemm_f16_wmma<<<gq, 256, 0, stream>>>(hoh, Hd, wph, Hd, Hd, nullptr, out, bp, Hd);
}
